// HGNNPConv_90022514524572
// MI455X (gfx1250) — matrix operand order repaired and checked by reading, not compiled, not run on hardware
//
#include <hip/hip_runtime.h>
#include <hip/hip_bf16.h>
#include <math.h>

#define NV   8192
#define NE   2048
#define CIN  128
#define COUT 128
#define NH   4
#define HID  32
#define EPSN 1e-5f

typedef __bf16 bf16;
typedef __attribute__((ext_vector_type(16))) __bf16 v16bf;
typedef __attribute__((ext_vector_type(8)))  float  v8f;
typedef __attribute__((ext_vector_type(4)))  unsigned int v4u;
typedef __attribute__((ext_vector_type(8)))  int v8i;
typedef __attribute__((ext_vector_type(4)))  int v4i;

#if __has_builtin(__builtin_amdgcn_tensor_load_to_lds) && __has_builtin(__builtin_amdgcn_s_wait_tensorcnt)
#define HAVE_TDM 1
#else
#define HAVE_TDM 0
#endif

__device__ __forceinline__ v8f wmma_bf16(v16bf a, v16bf b, v8f c) {
  return __builtin_amdgcn_wmma_f32_16x16x32_bf16(false, a, false, b, (short)0, c, false, false);
}
// A-matrix 16x32 bf16 slot->K map (ISA 7.12.2): slots 0..7 -> K=8g+s, slots 8..15 -> K=16+8g+(s-8)
__device__ __forceinline__ int a_slot_k(int s, int g) {
  return (s < 8) ? (8 * g + s) : (16 + 8 * g + (s - 8));
}
__device__ __forceinline__ float gelu_f(float x) {
  return 0.5f * x * (1.0f + erff(x * 0.70710678118654752f));
}
__device__ __forceinline__ float sigm_f(float x) { return 1.0f / (1.0f + expf(-x)); }

#if HAVE_TDM
// Issue a 2-D TDM tile load of H (f32) -> LDS, with pad_amount=1 DWORD inserted
// every tile row (`pad_code` 3 = every 16 DW, 4 = every 32 DW) so the LDS tile
// lands with a conflict-free padded stride. Descriptor per CDNA5 ISA 8.3/8.4.
// clang-23 toolchain: 6-arg builtin (g0, g1, g2, g3, g4, cpol).
__device__ __forceinline__ void tdm_load_tile_f32(const float* gbase, unsigned lds_off,
                                                  unsigned cols, unsigned rows,
                                                  unsigned pad_code) {
  unsigned long long ga = (unsigned long long)(const void*)gbase;
  v4u g0;
  g0[0] = 1u;                                                   // count=1, user mode
  g0[1] = lds_off;                                              // lds_addr
  g0[2] = (unsigned)ga;                                         // global_addr[31:0]
  g0[3] = ((unsigned)(ga >> 32) & 0x01FFFFFFu) | 0x80000000u;   // global_addr[56:32] | type=2
  v8i g1;
  g1[0] = (int)((2u << 16) | (1u << 20) | (pad_code << 22));    // data_size=4B, pad_enable, pad_interval, pad_amount=1DW
  g1[1] = (int)(((unsigned)NE & 0xFFFFu) << 16);                // tensor_dim0[15:0]
  g1[2] = (int)((((unsigned)NE >> 16) & 0xFFFFu) | (((unsigned)NV & 0xFFFFu) << 16)); // dim0 hi | dim1 lo
  g1[3] = (int)((((unsigned)NV >> 16) & 0xFFFFu) | (cols << 16));                      // dim1 hi | tile_dim0
  g1[4] = (int)rows;                                            // tile_dim1 | tile_dim2=0
  g1[5] = (int)NE;                                              // tensor_dim0_stride[31:0]
  g1[6] = 0;                                                    // stride hi | dim1_stride lo
  g1[7] = 0;
  v4i gz4 = {0, 0, 0, 0};
  v8i gz8 = {0, 0, 0, 0, 0, 0, 0, 0};
  __builtin_amdgcn_tensor_load_to_lds(g0, g1, gz4, gz4, gz8, 0);
  __builtin_amdgcn_s_wait_tensorcnt(0);
}
#endif

// ---------------- K1: per-head vertex attention MLP (VALU; 0.27 GFLOP) ------
__global__ void k_attn(const float* __restrict__ X, const float* __restrict__ vW1,
                       const float* __restrict__ vb1, const float* __restrict__ vW2,
                       const float* __restrict__ vb2, float* __restrict__ attn) {
  __shared__ float w1[CIN * HID];
  int v = blockIdx.x * 256 + threadIdx.x;
  for (int h = 0; h < NH; ++h) {
    __syncthreads();
    for (int i = threadIdx.x; i < CIN * HID; i += 256) w1[i] = vW1[h * CIN * HID + i];
    __syncthreads();
    float acc[HID];
#pragma unroll
    for (int k = 0; k < HID; ++k) acc[k] = vb1[h * HID + k];
    for (int c = 0; c < CIN; ++c) {
      float x = X[(size_t)v * CIN + c];
      const float* wr = &w1[c * HID];
#pragma unroll
      for (int k = 0; k < HID; ++k) acc[k] += x * wr[k];
    }
    float s = 0.f;
#pragma unroll
    for (int k = 0; k < HID; ++k) s += gelu_f(acc[k]) * vW2[h * HID + k];
    attn[(size_t)h * NV + v] = sigm_f(s + vb2[h]);
  }
}

// ---------------- K2: Xt = X@Wv+bv and GX = X@Wg_top (WMMA); emit bf16 WXt[h]
__global__ void k_xt(const float* __restrict__ X, const float* __restrict__ Wv,
                     const float* __restrict__ bv, const float* __restrict__ Wg,
                     const float* __restrict__ attn, float* __restrict__ Xt,
                     float* __restrict__ GX, bf16* __restrict__ WXt) {
  int wave = threadIdx.x >> 5, lane = threadIdx.x & 31;
  int g = lane >> 4, mr = lane & 15;
  int m0 = blockIdx.x * 128 + wave * 16;
  int krow = (lane & 15) + 16 * g;
  v16bf A[4];
#pragma unroll
  for (int ks = 0; ks < 4; ++ks)
#pragma unroll
    for (int s = 0; s < 16; ++s)
      A[ks][s] = (bf16)X[(size_t)(m0 + mr) * CIN + ks * 32 + a_slot_k(s, g)];
  for (int nt = 0; nt < 8; ++nt) {
    v8f accv = {}, accg = {};
#pragma unroll
    for (int ks = 0; ks < 4; ++ks) {
      v16bf Bv, Bg;
      const float* wv = Wv + (size_t)(ks * 32) * COUT + nt * 16 + (lane & 15);
      const float* wg = Wg + (size_t)(ks * 32) * COUT + nt * 16 + (lane & 15);
#pragma unroll
      for (int s = 0; s < 16; ++s) { Bv[s] = (bf16)wv[a_slot_k(s, g) * COUT]; Bg[s] = (bf16)wg[a_slot_k(s, g) * COUT]; }
      accv = wmma_bf16(A[ks], Bv, accv);
      accg = wmma_bf16(A[ks], Bg, accg);
    }
#pragma unroll
    for (int r = 0; r < 8; ++r) {
      int m = m0 + r + 8 * g, c = nt * 16 + (lane & 15);
      float xv = accv[r] + bv[c];
      Xt[(size_t)m * COUT + c] = xv;
      GX[(size_t)m * COUT + c] = accg[r];
#pragma unroll
      for (int h = 0; h < NH; ++h)
        WXt[((size_t)h * NV + m) * COUT + c] = (bf16)(xv * attn[(size_t)h * NV + m]);
    }
  }
}

// ---------------- K3: edge degrees + edge_att (last head) -------------------
__global__ void k_deg_att(const float* __restrict__ H, const float* __restrict__ attn,
                          float* __restrict__ deg, float* __restrict__ out_att) {
  int e = blockIdx.x * 256 + threadIdx.x;
  const float* a3 = attn + (size_t)(NH - 1) * NV;
  float d = 0.f, s = 0.f;
  for (int v = 0; v < NV; ++v) {
    if ((v & 63) == 0 && v + 128 < NV)
      __builtin_prefetch(H + (size_t)(v + 128) * NE + e, 0, 0);   // global_prefetch_b8
    float h = H[(size_t)v * NE + e];
    d += h; s += h * a3[v];
  }
  deg[e] = d;
  out_att[e] = s / fmaxf(d, 1.f);
}

// ---------------- K4: head_e = Ht @ WXt[h] / denom  (big H pass, TDM + WMMA)
// block: 16 edges; 8 waves = (head, col-half); H tile (32v x 16e f32) staged in
// LDS by the Tensor Data Mover with 1-DWORD row padding (bank-conflict free).
__global__ void k_head_e(const float* __restrict__ H, const bf16* __restrict__ WXt,
                         const float* __restrict__ deg, bf16* __restrict__ ecat) {
  __shared__ float Atf[32 * 17];               // [k][e], stride 17 (TDM pad)
  int tid = threadIdx.x, lane = tid & 31, wave = tid >> 5;
  int h = wave >> 1, nh = wave & 1;
  int g = lane >> 4, mr = lane & 15;
  int e0 = blockIdx.x * 16;
  int krow = (lane & 15) + 16 * g;
  int km[16];
#pragma unroll
  for (int s = 0; s < 16; ++s) km[s] = a_slot_k(s, g);
#if HAVE_TDM
  bool leader = (__builtin_amdgcn_readfirstlane((int)(tid >> 5)) == 0);
  unsigned lds_off = (unsigned)(size_t)(void*)Atf;
#endif
  v8f acc0 = {}, acc1 = {}, acc2 = {}, acc3 = {};
  for (int kb = 0; kb < NV; kb += 32) {
    __syncthreads();                           // WAR on the LDS tile
#if HAVE_TDM
    if (leader)
      tdm_load_tile_f32(H + (size_t)kb * NE + e0, lds_off,
                        /*cols=*/16, /*rows=*/32, /*pad: every 16 DW*/3u);
#else
    for (int i = tid; i < 512; i += 256) {
      int kk = i >> 4, ee = i & 15;
      Atf[kk * 17 + ee] = H[(size_t)(kb + kk) * NE + e0 + ee];
    }
#endif
    __syncthreads();
    v16bf a;
#pragma unroll
    for (int s = 0; s < 16; ++s) a[s] = (bf16)Atf[km[s] * 17 + mr];
    const bf16* brow = WXt + ((size_t)h * NV + kb) * COUT + nh * 64 + (lane & 15);
    v16bf b0, b1, b2, b3;
#pragma unroll
    for (int s = 0; s < 16; ++s) {
      const bf16* p = brow + (size_t)km[s] * COUT;
      b0[s] = p[0]; b1[s] = p[16]; b2[s] = p[32]; b3[s] = p[48];
    }
    acc0 = wmma_bf16(a, b0, acc0);
    acc1 = wmma_bf16(a, b1, acc1);
    acc2 = wmma_bf16(a, b2, acc2);
    acc3 = wmma_bf16(a, b3, acc3);
  }
  v8f accs[4] = {acc0, acc1, acc2, acc3};
#pragma unroll
  for (int t = 0; t < 4; ++t)
#pragma unroll
    for (int r = 0; r < 8; ++r) {
      int e = e0 + r + 8 * g;
      int n = nh * 64 + t * 16 + (lane & 15);
      float d = fmaxf(deg[e], 1.f);
      ecat[(size_t)e * (NH * COUT) + h * COUT + n] = (bf16)(accs[t][r] / d);
    }
}

// ---------------- K5: ef0 = ecat@Wc+bc ; te = EF@We+be (WMMA) ---------------
__global__ void k_edge(const bf16* __restrict__ ecat, const float* __restrict__ Wc,
                       const float* __restrict__ bc, const float* __restrict__ EF,
                       const float* __restrict__ We, const float* __restrict__ be,
                       float* __restrict__ ef0, float* __restrict__ tef) {
  int lane = threadIdx.x & 31, nt = threadIdx.x >> 5;
  int g = lane >> 4, mr = lane & 15;
  int e0 = blockIdx.x * 16;
  int krow = (lane & 15) + 16 * g;
  v8f acc = {};
  for (int kb = 0; kb < NH * COUT; kb += 32) {
    v16bf a, b;
#pragma unroll
    for (int s = 0; s < 16; ++s)
      a[s] = ecat[(size_t)(e0 + mr) * (NH * COUT) + kb + a_slot_k(s, g)];
    const float* w = Wc + (size_t)kb * COUT + nt * 16 + (lane & 15);
#pragma unroll
    for (int s = 0; s < 16; ++s) b[s] = (bf16)w[a_slot_k(s, g) * COUT];
    acc = wmma_bf16(a, b, acc);
  }
  v8f acct = {};
  for (int kb = 0; kb < COUT; kb += 32) {
    v16bf a, b;
#pragma unroll
    for (int s = 0; s < 16; ++s)
      a[s] = (bf16)EF[(size_t)(e0 + mr) * COUT + kb + a_slot_k(s, g)];
    const float* w = We + (size_t)kb * COUT + nt * 16 + (lane & 15);
#pragma unroll
    for (int s = 0; s < 16; ++s) b[s] = (bf16)w[a_slot_k(s, g) * COUT];
    acct = wmma_bf16(a, b, acct);
  }
#pragma unroll
  for (int r = 0; r < 8; ++r) {
    int e = e0 + r + 8 * g, c = nt * 16 + (lane & 15);
    ef0[(size_t)e * COUT + c] = acc[r] + bc[c];
    tef[(size_t)e * COUT + c] = acct[r] + be[c];
  }
}

// ---------------- K6: edge MLP gate + mix + per-row LayerNorm ---------------
__global__ void k_edge2(const float* __restrict__ ef0, const float* __restrict__ tef,
                        const float* __restrict__ eW1, const float* __restrict__ eb1,
                        const float* __restrict__ eW2, const float* __restrict__ eb2,
                        const float* __restrict__ lng, const float* __restrict__ lnb,
                        float* __restrict__ lnws) {
  __shared__ float w1[COUT * HID];
  __shared__ float sb1[HID], sw2[HID];
  for (int i = threadIdx.x; i < COUT * HID; i += 256) w1[i] = eW1[i];
  if (threadIdx.x < HID) { sb1[threadIdx.x] = eb1[threadIdx.x]; sw2[threadIdx.x] = eW2[threadIdx.x]; }
  __syncthreads();
  int e = blockIdx.x * 256 + threadIdx.x;
  const float* row = ef0 + (size_t)e * COUT;
  const float* trw = tef + (size_t)e * COUT;
  float sc = eb2[0];
  for (int k = 0; k < HID; ++k) {
    float acc = sb1[k];
    for (int c = 0; c < COUT; ++c) acc += row[c] * w1[c * HID + k];
    sc += gelu_f(acc) * sw2[k];
  }
  float ew = sigm_f(sc);
  float mix[COUT];
  float mean = 0.f;
  for (int c = 0; c < COUT; ++c) { mix[c] = row[c] * ew + trw[c] * (1.f - ew); mean += mix[c]; }
  mean *= (1.f / COUT);
  float var = 0.f;
  for (int c = 0; c < COUT; ++c) { float d = mix[c] - mean; var += d * d; }
  var *= (1.f / COUT);
  float inv = rsqrtf(var + EPSN);
  for (int c = 0; c < COUT; ++c)
    lnws[(size_t)e * COUT + c] = (mix[c] - mean) * inv * lng[c] + lnb[c];
}

// ---------------- K7: deterministic BN column stats -------------------------
__global__ void k_bnstats(const float* __restrict__ lnws, float* __restrict__ bn) {
  int c = threadIdx.x;  // 128 threads
  float s = 0.f, q = 0.f;
  for (int e = 0; e < NE; ++e) {
    float x = lnws[(size_t)e * COUT + c];
    s += x; q += x * x;
  }
  bn[c] = s; bn[COUT + c] = q;
}

// ---------------- K8: apply BN; emit bf16 e_feat + gelu(e_feat) out ---------
__global__ void k_bn(const float* __restrict__ lnws, const float* __restrict__ bn,
                     const float* __restrict__ bng, const float* __restrict__ bnb,
                     bf16* __restrict__ efbf, float* __restrict__ out_e) {
  int i = blockIdx.x * 256 + threadIdx.x;
  int c = i & (COUT - 1);
  float m = bn[c] * (1.f / NE);
  float var = bn[COUT + c] * (1.f / NE) - m * m;
  float y = (lnws[i] - m) * rsqrtf(var + EPSN) * bng[c] + bnb[c];
  efbf[i] = (bf16)y;
  out_e[i] = gelu_f(y);
}

// ---------------- K9: v_feat = H @ e_feat (second H pass, TDM + WMMA) -------
__global__ void k_vfeat(const float* __restrict__ H, const bf16* __restrict__ efbf,
                        float* __restrict__ vf, bf16* __restrict__ vfbf) {
  __shared__ float Arf[16 * 33];               // [v][k], stride 33 (TDM pad)
  int tid = threadIdx.x, lane = tid & 31, nt = tid >> 5;
  int g = lane >> 4, mr = lane & 15;
  int v0 = blockIdx.x * 16;
  int krow = (lane & 15) + 16 * g;
  int km[16];
#pragma unroll
  for (int s = 0; s < 16; ++s) km[s] = a_slot_k(s, g);
#if HAVE_TDM
  bool leader = (__builtin_amdgcn_readfirstlane((int)(tid >> 5)) == 0);
  unsigned lds_off = (unsigned)(size_t)(void*)Arf;
#endif
  v8f acc = {};
  for (int kb = 0; kb < NE; kb += 32) {
    __syncthreads();
#if HAVE_TDM
    if (leader)
      tdm_load_tile_f32(H + (size_t)v0 * NE + kb, lds_off,
                        /*cols=*/32, /*rows=*/16, /*pad: every 32 DW*/4u);
#else
    for (int i = tid; i < 512; i += 256) {
      int rr = i >> 5, kk = i & 31;
      Arf[rr * 33 + kk] = H[(size_t)(v0 + rr) * NE + kb + kk];
    }
#endif
    __syncthreads();
    v16bf a;
#pragma unroll
    for (int s = 0; s < 16; ++s) a[s] = (bf16)Arf[mr * 33 + km[s]];
    v16bf b;
#pragma unroll
    for (int s = 0; s < 16; ++s) b[s] = efbf[(size_t)(kb + km[s]) * COUT + nt * 16 + (lane & 15)];
    acc = wmma_bf16(a, b, acc);
  }
#pragma unroll
  for (int r = 0; r < 8; ++r) {
    int m = v0 + r + 8 * g, c = nt * 16 + (lane & 15);
    vf[(size_t)m * COUT + c] = acc[r];
    vfbf[(size_t)m * COUT + c] = (bf16)acc[r];
  }
}

// ---------------- K10: GV = v_feat @ Wg_bot (WMMA) --------------------------
__global__ void k_gv(const bf16* __restrict__ vfbf, const float* __restrict__ Wg,
                     float* __restrict__ GV) {
  int wave = threadIdx.x >> 5, lane = threadIdx.x & 31;
  int g = lane >> 4, mr = lane & 15;
  int m0 = blockIdx.x * 128 + wave * 16;
  int krow = (lane & 15) + 16 * g;
  const float* WgB = Wg + (size_t)CIN * COUT;
  v16bf A[4];
#pragma unroll
  for (int ks = 0; ks < 4; ++ks)
#pragma unroll
    for (int s = 0; s < 16; ++s)
      A[ks][s] = vfbf[(size_t)(m0 + mr) * COUT + ks * 32 + a_slot_k(s, g)];
  for (int nt = 0; nt < 8; ++nt) {
    v8f acc = {};
#pragma unroll
    for (int ks = 0; ks < 4; ++ks) {
      v16bf B;
      const float* w = WgB + (size_t)(ks * 32) * COUT + nt * 16 + (lane & 15);
#pragma unroll
      for (int s = 0; s < 16; ++s) B[s] = (bf16)w[a_slot_k(s, g) * COUT];
      acc = wmma_bf16(A[ks], B, acc);
    }
#pragma unroll
    for (int r = 0; r < 8; ++r)
      GV[(size_t)(m0 + r + 8 * g) * COUT + nt * 16 + (lane & 15)] = acc[r];
  }
}

// ---------------- K11: gate + mix + LN + residual + gelu (wave per vertex) --
__global__ void k_vout(const float* __restrict__ vf, const float* __restrict__ GX,
                       const float* __restrict__ GV, const float* __restrict__ bg,
                       const float* __restrict__ Xt, const float* __restrict__ X,
                       const float* __restrict__ lvg, const float* __restrict__ lvb,
                       float* __restrict__ out_v) {
  int wave = threadIdx.x >> 5, lane = threadIdx.x & 31;
  int v = blockIdx.x * 8 + wave;
  float val[4], s = 0.f, s2 = 0.f;
#pragma unroll
  for (int j = 0; j < 4; ++j) {
    int c = lane + 32 * j;
    size_t i = (size_t)v * COUT + c;
    float gate = sigm_f(GX[i] + GV[i] + bg[c]);
    float t = gate * vf[i] + (1.f - gate) * Xt[i];
    val[j] = t; s += t; s2 += t * t;
  }
  for (int off = 16; off > 0; off >>= 1) {
    s += __shfl_xor(s, off, 32);
    s2 += __shfl_xor(s2, off, 32);
  }
  float mean = s * (1.f / COUT);
  float var = s2 * (1.f / COUT) - mean * mean;
  float inv = rsqrtf(var + EPSN);
#pragma unroll
  for (int j = 0; j < 4; ++j) {
    int c = lane + 32 * j;
    size_t i = (size_t)v * COUT + c;
    float y = (val[j] - mean) * inv * lvg[c] + lvb[c] + X[i];
    out_v[i] = gelu_f(y);
  }
}

extern "C" void kernel_launch(void* const* d_in, const int* in_sizes, int n_in,
                              void* d_out, int out_size, void* d_ws, size_t ws_size,
                              hipStream_t stream) {
  const float* X   = (const float*)d_in[0];
  const float* H   = (const float*)d_in[1];
  const float* EF  = (const float*)d_in[2];
  const float* Wv  = (const float*)d_in[3];
  const float* bv  = (const float*)d_in[4];
  const float* We  = (const float*)d_in[5];
  const float* be  = (const float*)d_in[6];
  const float* vW1 = (const float*)d_in[7];
  const float* vb1 = (const float*)d_in[8];
  const float* vW2 = (const float*)d_in[9];
  const float* vb2 = (const float*)d_in[10];
  const float* eW1 = (const float*)d_in[11];
  const float* eb1 = (const float*)d_in[12];
  const float* eW2 = (const float*)d_in[13];
  const float* eb2 = (const float*)d_in[14];
  const float* Wc  = (const float*)d_in[15];
  const float* bc  = (const float*)d_in[16];
  const float* lng = (const float*)d_in[17];
  const float* lnb = (const float*)d_in[18];
  const float* lvg = (const float*)d_in[19];
  const float* lvb = (const float*)d_in[20];
  const float* bng = (const float*)d_in[21];
  const float* bnb = (const float*)d_in[22];
  const float* Wg  = (const float*)d_in[23];
  const float* bg  = (const float*)d_in[24];

  float* out_v   = (float*)d_out;                       // (V, COUT)
  float* out_e   = out_v + (size_t)NV * COUT;           // (E, COUT)
  float* out_att = out_e + (size_t)NE * COUT;           // (E,)

  char* w = (char*)d_ws;
  size_t off = 0;
  auto grab = [&](size_t bytes) { char* p = w + off; off = (off + bytes + 255) & ~(size_t)255; return p; };
  float* attn  = (float*)grab((size_t)NH * NV * 4);
  float* deg   = (float*)grab((size_t)NE * 4);
  float* Xt    = (float*)grab((size_t)NV * COUT * 4);
  float* GX    = (float*)grab((size_t)NV * COUT * 4);
  bf16*  WXt   = (bf16*) grab((size_t)NH * NV * COUT * 2);
  bf16*  ecat  = (bf16*) grab((size_t)NE * NH * COUT * 2);
  float* ef0   = (float*)grab((size_t)NE * COUT * 4);
  float* tef   = (float*)grab((size_t)NE * COUT * 4);
  float* lnws  = (float*)grab((size_t)NE * COUT * 4);
  float* bnst  = (float*)grab((size_t)2 * COUT * 4);
  bf16*  efbf  = (bf16*) grab((size_t)NE * COUT * 2);
  float* vf    = (float*)grab((size_t)NV * COUT * 4);
  bf16*  vfbf  = (bf16*) grab((size_t)NV * COUT * 2);
  float* GV    = (float*)grab((size_t)NV * COUT * 4);
  (void)ws_size; (void)n_in; (void)in_sizes; (void)out_size;

  k_attn   <<<NV / 256, 256, 0, stream>>>(X, vW1, vb1, vW2, vb2, attn);
  k_xt     <<<NV / 128, 256, 0, stream>>>(X, Wv, bv, Wg, attn, Xt, GX, WXt);
  k_deg_att<<<NE / 256, 256, 0, stream>>>(H, attn, deg, out_att);
  k_head_e <<<NE / 16, 256, 0, stream>>>(H, WXt, deg, ecat);
  k_edge   <<<NE / 16, 256, 0, stream>>>(ecat, Wc, bc, EF, We, be, ef0, tef);
  k_edge2  <<<NE / 256, 256, 0, stream>>>(ef0, tef, eW1, eb1, eW2, eb2, lng, lnb, lnws);
  k_bnstats<<<1, COUT, 0, stream>>>(lnws, bnst);
  k_bn     <<<(NE * COUT) / 256, 256, 0, stream>>>(lnws, bnst, bng, bnb, efbf, out_e);
  k_vfeat  <<<NV / 16, 256, 0, stream>>>(H, efbf, vf, vfbf);
  k_gv     <<<NV / 128, 256, 0, stream>>>(vfbf, Wg, GV);
  k_vout   <<<NV / 8, 256, 0, stream>>>(vf, GX, GV, bg, Xt, X, lvg, lvb, out_v);
}
